// CIDL_RNN_18897856103013
// MI455X (gfx1250) — matrix operand order repaired and checked by reading, not compiled, not run on hardware
//
#include <hip/hip_runtime.h>
#include <hip/hip_bf16.h>

// CIDL-LSTM for MI455X (gfx1250), wave32 + WMMA.
// B=32, T=8192, I=1, H=64.  2 workgroups x 16 waves; per step:
//   gates(16x256) = A(16x64 f16 h_read) @ B(64x256 f16 W_hh^T) + gx_t  via
//   2 chained v_wmma_f32_16x16x32_f16 per wave, then in-wave gate gather
//   (v_mov_b32_dpp8 / v_permlane16, VALU-only) and register-resident cubic
//   window update. 1 barrier/step, branchless stores, time loop unrolled x4.

typedef __attribute__((ext_vector_type(16))) _Float16 v16h;
typedef __attribute__((ext_vector_type(8)))  _Float16 v8h;
typedef __attribute__((ext_vector_type(8)))  float    v8f;

#define TSEQ 8192
#define HID  64
#define ASTR 72   // Alds row stride in halfs (144B, 16B-multiple, bank-staggered)

// cubic kernel for OS_FACTOR=1.5 -> delta=0.5
#define K0 0.3125f
#define K1 0.9375f
#define K2 (-0.3125f)
#define K3 0.0625f

__device__ __forceinline__ float shfl32(float v, int srcLane) {
  return __int_as_float(
      __builtin_amdgcn_ds_bpermute(srcLane << 2, __float_as_int(v)));
}

// Gather the gate value held by lane ((lane & ~3) | P) within each 4-lane
// group (groups never cross 8-lane DPP rows / 16-lane halves).
template <int P>
__device__ __forceinline__ float gather_gate(float v, int base) {
#if __has_builtin(__builtin_amdgcn_mov_dpp8)
  // DPP8: lane i reads lane (i&~3)|P of its 8-lane row; no tied old operand.
  const int sel = (int)((unsigned)(P * 0x249u) + (unsigned)((4 + P) * 0x249000u));
  return __int_as_float(__builtin_amdgcn_mov_dpp8(__float_as_int(v), sel));
#elif __has_builtin(__builtin_amdgcn_permlane16)
  const unsigned sel0 = (unsigned)(P * 0x1111) | (unsigned)((4 + P) * 0x11110000u);
  const unsigned sel1 = (unsigned)((8 + P) * 0x1111) | (unsigned)((12 + P) * 0x11110000u);
  unsigned s = (unsigned)__float_as_int(v);
  return __int_as_float(
      (int)__builtin_amdgcn_permlane16(s, s, sel0, sel1, false, false));
#else
  return shfl32(v, base + P);
#endif
}

// fast tanh: single v_tanh_f32 when available, else exp + v_rcp_f32
__device__ __forceinline__ float tanh_hw(float x) {
#if __has_builtin(__builtin_amdgcn_tanhf)
  return __builtin_amdgcn_tanhf(x);
#else
  float e2 = __expf(-2.f * x);
  return (1.f - e2) * __builtin_amdgcn_rcpf(1.f + e2);
#endif
}

__global__ __launch_bounds__(512) void cidl_lstm_kernel(
    const float* __restrict__ x,     // (32, 8192, 1)
    const float* __restrict__ W_ih,  // (256, 1)
    const float* __restrict__ W_hh,  // (256, 64)
    const float* __restrict__ b_ih,  // (256,)
    const float* __restrict__ b_hh,  // (256,)
    float* __restrict__ out)         // h_all | h_all | c_all
{
  __shared__ __align__(16) _Float16 Alds[2][16 * ASTR]; // h_read f16, double-buffered
  __shared__ __align__(16) float    xsb[2][32][16];     // x stream, 32-step blocks

  const int tid  = threadIdx.x;
  const int w    = tid >> 5;        // wave id: N-tile
  const int lane = tid & 31;
  const int nl   = lane & 15;
  const int hi   = lane >> 4;       // half-wave
  const int p4   = lane & 3;        // position within 4-lane gate group
  const int bg0  = blockIdx.x * 16; // first batch row of this WG

  // Gate-major permutation: column Np = j*4 + gate  (gate: 0=i,1=f,2=g,3=o)
  const int Np   = 16 * w + nl;     // this lane's output column (C layout: N=lane%16)
  const int j    = Np >> 2;         // hidden unit
  const int g    = Np & 3;          // gate id
  const int orig = g * 64 + j;      // row in original [i|f|g|o] layout

  const float wih  = W_ih[orig];
  const float bsum = b_ih[orig] + b_hh[orig];

  // ---- preload B operand (W_hh^T tile) as f16, ISA dense-B layout:
  // lanes 0-15 hold K={0..7,16..23}, lanes 16-31 hold K={8..15,24..31}, column = lane%16.
  const int kb = hi * 8;
  v16h b0, b1;
  {
    const float* wr = W_hh + (size_t)orig * HID + kb;
#pragma unroll
    for (int e = 0; e < 16; ++e) b0[e] = (_Float16)wr[e < 8 ? e : e + 8];
    const float* wr2 = wr + 32;
#pragma unroll
    for (int e = 0; e < 16; ++e) b1[e] = (_Float16)wr2[e < 8 ? e : e + 8];
  }

  // ---- init LDS: A staging zero (win0 == 0), first x block
  for (int i = tid; i < 2 * 16 * ASTR; i += 512)
    ((_Float16*)Alds)[i] = (_Float16)0.0f;
  {
    const int bb = tid >> 5, tt = tid & 31;
    xsb[0][tt][bb] = x[(size_t)(bg0 + bb) * TSEQ + tt];
  }

  // ---- rolling per-(M=r+hi*8, j) state in registers (redundant across the
  // 4-lane gate group; same values in all 4 lanes).
  float h1[8], h2[8], h3[8], c1[8], c2[8], c3[8], crd[8];
#pragma unroll
  for (int r = 0; r < 8; ++r) {
    h1[r] = h2[r] = h3[r] = 0.f;
    c1[r] = c2[r] = c3[r] = 0.f;
    crd[r] = 0.f;   // c_read(t=0) = 0
  }

  // ---- branchless output routing:
  //   g==0 -> h copy 1, g==1 -> h copy 2, g==2 and g==3 -> c (duplicate
  //   store of identical redundant value; deterministic).
  //   Two running pointers; per-row offsets r*TSEQ*HID*4B (0..6MB) fold into
  //   the 24-bit signed instruction offset.
  float* o1 = out;
  float* o2 = out + (size_t)32 * TSEQ * HID;
  float* oc = o2 + (size_t)32 * TSEQ * HID;
  float* obase = (g == 0) ? o1 : (g == 1) ? o2 : oc;
  float* prlo = obase + (size_t)(bg0 + 0 + hi * 8) * TSEQ * HID + j; // rows 0-3
  float* prhi = obase + (size_t)(bg0 + 4 + hi * 8) * TSEQ * HID + j; // rows 4-7

  // A-staging store slots: the 4 group lanes each store rows {2p, 2p+1}
  // (values are identical across the group), so no exec toggle is needed.
  const int aw0 = (2 * p4 + hi * 8) * ASTR + j;

  float xfut = 0.f;

#pragma unroll 4
  for (int t = 0; t < TSEQ; ++t) {
    __syncthreads();  // Alds[ab] and xsb block for t are ready
    const int ab = t & 1;

    // issue next 32-step x block load early (latency hidden behind compute)
    const int tb = (t & ~31) + 32;
    if ((t & 31) == 0 && tb < TSEQ) {
      const int bb = tid >> 5, tt = tid & 31;
      xfut = x[(size_t)(bg0 + bb) * TSEQ + tb + tt];
    }

    // ---- load A operand (h_read, 16x64 f16) per ISA 16-bit A layout:
    // lane<16: M=lane, K in {0..7,16..23}; lane>=16: M=lane-16, K in {8..15,24..31}
    const _Float16* Ab = Alds[ab];
    const int c0 = hi * 8;
    v8h l0 = *(const v8h*)&Ab[nl * ASTR + c0];
    v8h m0 = *(const v8h*)&Ab[nl * ASTR + c0 + 16];
    v8h l1 = *(const v8h*)&Ab[nl * ASTR + 32 + c0];
    v8h m1 = *(const v8h*)&Ab[nl * ASTR + 32 + c0 + 16];
    v16h a0 = __builtin_shufflevector(l0, m0, 0,1,2,3,4,5,6,7,8,9,10,11,12,13,14,15);
    v16h a1 = __builtin_shufflevector(l1, m1, 0,1,2,3,4,5,6,7,8,9,10,11,12,13,14,15);

    // ---- C init = gx_t + biases:  C[r] -> row M=r+hi*8, col Np
    const float* xr = &xsb[(t >> 5) & 1][t & 31][hi * 8];
    float4 xa = *(const float4*)xr;
    float4 xc = *(const float4*)(xr + 4);
    v8f acc;
    acc[0] = __builtin_fmaf(xa.x, wih, bsum);
    acc[1] = __builtin_fmaf(xa.y, wih, bsum);
    acc[2] = __builtin_fmaf(xa.z, wih, bsum);
    acc[3] = __builtin_fmaf(xa.w, wih, bsum);
    acc[4] = __builtin_fmaf(xc.x, wih, bsum);
    acc[5] = __builtin_fmaf(xc.y, wih, bsum);
    acc[6] = __builtin_fmaf(xc.z, wih, bsum);
    acc[7] = __builtin_fmaf(xc.w, wih, bsum);

    // ---- gates = A x B + C   (K=64 -> two chained WMMAs)
    acc = __builtin_amdgcn_wmma_f32_16x16x32_f16(false, a0, false, b0,
                                                 (short)0, acc, false, false);
    acc = __builtin_amdgcn_wmma_f32_16x16x32_f16(false, a1, false, b1,
                                                 (short)0, acc, false, false);

    // ---- activation via a single tanh each:
    //   g-gate:   tanh(x)
    //   i/f/o:    sigmoid(x) = 0.5*tanh(0.5*x) + 0.5
    const float sc = (g == 2) ? 1.f : 0.5f;
    const float sb = (g == 2) ? 0.f : 0.5f;
    float act[8];
#pragma unroll
    for (int r = 0; r < 8; ++r) {
      float th = tanh_hw(sc * acc[r]);
      act[r]   = __builtin_fmaf(sc, th, sb);
    }

    // ---- gather i,f,g,o within the 4-lane group; LSTM cell + cubic window
    const int base = lane & ~3;
    _Float16* Aw = Alds[ab ^ 1];
    float hrf[8];
#pragma unroll
    for (int r = 0; r < 8; ++r) {
      float iv = gather_gate<0>(act[r], base);
      float fv = gather_gate<1>(act[r], base);
      float gv = gather_gate<2>(act[r], base);
      float ov = gather_gate<3>(act[r], base);

      float cn = __builtin_fmaf(fv, crd[r], iv * gv);
      float hn = ov * tanh_hw(cn);

      // next step's interpolated reads (needs states t .. t-3)
      hrf[r]    = K0 * hn + K1 * h1[r] + K2 * h2[r] + K3 * h3[r];
      float cr2 = K0 * cn + K1 * c1[r] + K2 * c2[r] + K3 * c3[r];
      h3[r] = h2[r]; h2[r] = h1[r]; h1[r] = hn;
      c3[r] = c2[r]; c2[r] = c1[r]; c1[r] = cn; crd[r] = cr2;

      // branchless routed store (g0/g1: h, g2/g3: c duplicate);
      // row offset folds into the 24-bit instruction immediate.
      float vsel = (g >= 2) ? cn : hn;
      float* pp = (r < 4) ? prlo : prhi;
      __builtin_nontemporal_store(vsel, pp + (size_t)(r & 3) * TSEQ * HID);
    }
    prlo += HID;
    prhi += HID;

    // A staging for step t+1: group lane p stores rows {2p, 2p+1}
    // (identical redundant values -> any lane may store any row).
    {
      float e0 = (p4 == 0) ? hrf[0] : (p4 == 1) ? hrf[2] : (p4 == 2) ? hrf[4] : hrf[6];
      float e1 = (p4 == 0) ? hrf[1] : (p4 == 1) ? hrf[3] : (p4 == 2) ? hrf[5] : hrf[7];
      Aw[aw0]        = (_Float16)e0;
      Aw[aw0 + ASTR] = (_Float16)e1;
    }

    // ---- commit next x block into the other buffer
    if ((t & 31) == 0 && tb < TSEQ) {
      const int bb = tid >> 5, tt = tid & 31;
      xsb[((t >> 5) & 1) ^ 1][tt][bb] = xfut;
    }
  }
}

extern "C" void kernel_launch(void* const* d_in, const int* in_sizes, int n_in,
                              void* d_out, int out_size, void* d_ws, size_t ws_size,
                              hipStream_t stream) {
  (void)in_sizes; (void)n_in; (void)out_size; (void)d_ws; (void)ws_size;
  const float* x    = (const float*)d_in[0];
  const float* W_ih = (const float*)d_in[1];
  const float* W_hh = (const float*)d_in[2];
  const float* b_ih = (const float*)d_in[3];
  const float* b_hh = (const float*)d_in[4];
  float* out = (float*)d_out;
  cidl_lstm_kernel<<<2, 512, 0, stream>>>(x, W_ih, W_hh, b_ih, b_hh, out);
}
